// DynamicBlock_13280038879407
// MI455X (gfx1250) — matrix operand order repaired and checked by reading, not compiled, not run on hardware
//
#include <hip/hip_runtime.h>
#include <math.h>

// ---------------------------------------------------------------------------
// Problem constants (match reference)
// ---------------------------------------------------------------------------
constexpr int Bc = 4, Tc = 4096, Dc = 2048, Kc = 1024;
constexpr int Hc = 16, KVHc = 4, HDc = 128, FFc = 8192;
constexpr float EPSc = 1e-6f;
constexpr int NTOK = Bc * Kc;                  // 4096 selected tokens total
constexpr float QSCALE = 0.08838834764831845f; // 1/sqrt(128)

typedef __attribute__((ext_vector_type(16))) __bf16 v16bf;
typedef __attribute__((ext_vector_type(8)))  __bf16 v8bf;
typedef __attribute__((ext_vector_type(8)))  float  v8f;
typedef int v4i_ __attribute__((__vector_size__(16)));  // matches builtin param

__device__ __forceinline__ v16bf pack16(v8bf lo, v8bf hi) {
  v16bf r;
#pragma unroll
  for (int i = 0; i < 8; ++i) { r[i] = lo[i]; r[i + 8] = hi[i]; }
  return r;
}

// ---------------------------------------------------------------------------
// CDNA5 async global->LDS copy (ASYNCcnt path), with safe fallback.
// 16 bytes per lane per issue (GLOBAL_LOAD_ASYNC_TO_LDS_B128).
// ---------------------------------------------------------------------------
#if __has_builtin(__builtin_amdgcn_global_load_async_to_lds_b128)
#define HAVE_ASYNC_LDS 1
#else
#define HAVE_ASYNC_LDS 0
#endif

__device__ __forceinline__ void cp_async16(const __bf16* g, __bf16* l) {
#if HAVE_ASYNC_LDS
  __builtin_amdgcn_global_load_async_to_lds_b128(
      (__attribute__((address_space(1))) v4i_*)(uintptr_t)g,
      (__attribute__((address_space(3))) v4i_*)(uint32_t)(uintptr_t)l,
      0, 0);
#else
  *(v8bf*)l = *(const v8bf*)g;
#endif
}

template <int N>
__device__ __forceinline__ void wait_async() {
#if HAVE_ASYNC_LDS
#if __has_builtin(__builtin_amdgcn_s_wait_asynccnt)
  __builtin_amdgcn_s_wait_asynccnt(N);
#elif defined(__HIP_DEVICE_COMPILE__)
  asm volatile("s_wait_asynccnt %0" ::"i"(N));
#endif
#endif
}

// ---------------------------------------------------------------------------
// fp32 (K x N) -> bf16 transposed (N x K): weights become N-major so GEMM
// B-tiles are K-contiguous (identical staging to A, async-friendly).
// Tiled 32x32 transpose through LDS. Grid: (N/32, K/32), 256 threads.
// ---------------------------------------------------------------------------
__global__ __launch_bounds__(256)
void f2bT_kernel(const float* __restrict__ in, __bf16* __restrict__ out,
                 int Kd, int N) {
  __shared__ float t[32][33];
  const int kt = blockIdx.y * 32, nt = blockIdx.x * 32;
  const int tx = threadIdx.x & 31, ty = threadIdx.x >> 5;  // ty in 0..7
#pragma unroll
  for (int j = 0; j < 4; ++j)
    t[ty + j * 8][tx] = in[(size_t)(kt + ty + j * 8) * N + nt + tx];
  __syncthreads();
#pragma unroll
  for (int j = 0; j < 4; ++j)
    out[(size_t)(nt + ty + j * 8) * Kd + kt + tx] = (__bf16)t[tx][ty + j * 8];
}

// ---------------------------------------------------------------------------
// (gather +) RMSNorm: one block per token, 256 threads x 8 elems = D=2048
// ---------------------------------------------------------------------------
template <bool GATHER>
__global__ __launch_bounds__(256)
void rmsnorm_kernel(const float* __restrict__ xin, const long long* __restrict__ idx,
                    const float* __restrict__ w, float* __restrict__ selOut,
                    __bf16* __restrict__ hOut) {
  const int b = blockIdx.y, i = blockIdx.x, tid = threadIdx.x;
  const size_t tokrow = (size_t)(b * Kc + i);
  const float* row;
  if (GATHER) {
    const long long t = idx[b * Kc + i];
    row = xin + ((size_t)b * Tc + (size_t)t) * Dc;
  } else {
    row = xin + tokrow * Dc;
  }
  float vals[8];
  float ss = 0.f;
#pragma unroll
  for (int j = 0; j < 8; ++j) {
    float v = row[tid + j * 256];
    vals[j] = v;
    ss += v * v;
  }
#pragma unroll
  for (int off = 16; off >= 1; off >>= 1) ss += __shfl_xor(ss, off, 32);
  __shared__ float red[8];
  if ((tid & 31) == 0) red[tid >> 5] = ss;
  __syncthreads();
  float tot = 0.f;
#pragma unroll
  for (int j = 0; j < 8; ++j) tot += red[j];
  const float inv = rsqrtf(tot / (float)Dc + EPSc);
#pragma unroll
  for (int j = 0; j < 8; ++j) {
    const int c = tid + j * 256;
    if (GATHER) selOut[tokrow * Dc + c] = vals[j];
    hOut[tokrow * Dc + c] = (__bf16)(vals[j] * inv * w[c]);
  }
}

// ---------------------------------------------------------------------------
// bf16 WMMA GEMM: C[M,N] = A(MxK, row-major) x Bt(NxK, row-major = B^T)
// 128x128 block tile, K-step 32, double-buffered async LDS staging.
// 8 waves (4x2), each wave 2x4 16x16 WMMA tiles.
// EPI: 0 f32 | 1 f32+bias | 4 f32=resid+acc | 5 bf16=silu(gate)*acc
//      6 bf16+bias written transposed per (b,kvh): out[(b*KVH+kvh)*HD+hd][tok]
// ---------------------------------------------------------------------------
constexpr int GBM = 128, GBN = 128, GBK = 32;
constexpr int LDT_S = GBK + 8;  // padded tile row (elems), 80B (16B aligned)

template <int EPI>
__global__ __launch_bounds__(256)
void gemm_bf16(const __bf16* __restrict__ A, const __bf16* __restrict__ Bt,
               int N, int Kd,
               const float* __restrict__ bias, const float* __restrict__ resid,
               const float* __restrict__ gatebuf,
               float* __restrict__ outF, __bf16* __restrict__ outB) {
  __shared__ __bf16 As[2][GBM * LDT_S];
  __shared__ __bf16 Bs[2][GBN * LDT_S];
  const int bn = blockIdx.x, bm = blockIdx.y;
  const int tid = threadIdx.x;
  const int lane = tid & 31, wid = tid >> 5;
  const int half = lane >> 4, l15 = lane & 15;
  const int m0 = (wid >> 1) * 32, n0 = (wid & 1) * 64;
  v8f acc[2][4] = {};
  const __bf16* Ab = A + (size_t)bm * GBM * Kd;
  const __bf16* Bb = Bt + (size_t)bn * GBN * Kd;
  const int sr = tid >> 2, sc = (tid & 3) * 8;  // stage mapping: 64 rows x 4 chunks

  auto stage = [&](int k0, int buf) {
#pragma unroll
    for (int p = 0; p < 2; ++p) {
      const int row = sr + p * 64;
      cp_async16(Ab + (size_t)row * Kd + k0 + sc, &As[buf][row * LDT_S + sc]);
      cp_async16(Bb + (size_t)row * Kd + k0 + sc, &Bs[buf][row * LDT_S + sc]);
    }
  };

  stage(0, 0);
  const int nk = Kd / GBK;
  for (int kb = 0; kb < nk; ++kb) {
    const int cur = kb & 1;
    if (kb + 1 < nk) {
      stage((kb + 1) * GBK, cur ^ 1);
      wait_async<4>();  // in-order completion => tile kb landed; kb+1 in flight
    } else {
      wait_async<0>();
    }
    __syncthreads();
    v16bf af[2], bfg[4];
#pragma unroll
    for (int mt = 0; mt < 2; ++mt) {  // A frag: e<8 -> k=half*8+e ; e>=8 -> 16+half*8+(e-8)
      const __bf16* p = &As[cur][(m0 + mt * 16 + l15) * LDT_S];
      af[mt] = pack16(*(const v8bf*)(p + half * 8), *(const v8bf*)(p + 16 + half * 8));
    }
#pragma unroll
    for (int nt = 0; nt < 4; ++nt) {  // B frag: e<8 -> k=half*8+e ; e>=8 -> 16+half*8+(e-8)
      const __bf16* p = &Bs[cur][(n0 + nt * 16 + l15) * LDT_S];
      bfg[nt] = pack16(*(const v8bf*)(p + half * 8), *(const v8bf*)(p + 16 + half * 8));
    }
#pragma unroll
    for (int mt = 0; mt < 2; ++mt)
#pragma unroll
      for (int nt = 0; nt < 4; ++nt)
        acc[mt][nt] = __builtin_amdgcn_wmma_f32_16x16x32_bf16(
            false, af[mt], false, bfg[nt], (short)0, acc[mt][nt], false, false);
    __syncthreads();  // protect buffer 'cur' before it is restaged next+1 iter
  }

#pragma unroll
  for (int mt = 0; mt < 2; ++mt)
#pragma unroll
    for (int nt = 0; nt < 4; ++nt)
#pragma unroll
      for (int r = 0; r < 8; ++r) {
        const int grow = bm * GBM + m0 + mt * 16 + half * 8 + r;
        const int gcol = bn * GBN + n0 + nt * 16 + l15;
        const size_t o = (size_t)grow * N + gcol;
        float v = acc[mt][nt][r];
        if (EPI == 1 || EPI == 6) v += bias[gcol];
        if (EPI == 0 || EPI == 1) outF[o] = v;
        if (EPI == 4) outF[o] = resid[o] + v;
        if (EPI == 5) {
          float g = gatebuf[o];
          float s = g / (1.f + __expf(-g));
          outB[o] = (__bf16)(s * v);
        }
        if (EPI == 6) {  // V: write transposed [ (b*KVH+kvh)*HD + hd ][ tok ]
          const int bb = grow / Kc, ii = grow - bb * Kc;
          outB[((size_t)bb * (KVHc * HDc) + gcol) * Kc + ii] = (__bf16)v;
        }
      }
}

// ---------------------------------------------------------------------------
// RoPE (gathered cos/sin). SCALE folds 1/sqrt(HD) into Q. fp32 in, bf16 out.
// ---------------------------------------------------------------------------
template <bool SCALE>
__global__ __launch_bounds__(256)
void rope_kernel(const float* __restrict__ in, const float* __restrict__ cosT,
                 const float* __restrict__ sinT, const long long* __restrict__ idx,
                 __bf16* __restrict__ out, int nH) {
  const size_t gid = (size_t)blockIdx.x * 256 + threadIdx.x;
  const size_t total = (size_t)NTOK * nH * (HDc / 2);
  if (gid >= total) return;
  const int d = (int)(gid & 63);
  size_t rest = gid >> 6;
  const int hh = (int)(rest % nH);
  const size_t tok = rest / nH;
  const int b = (int)(tok / Kc), i = (int)(tok % Kc);
  const long long t = idx[b * Kc + i];
  const float* crow = cosT + ((size_t)b * Tc + (size_t)t) * HDc;
  const float* srow = sinT + ((size_t)b * Tc + (size_t)t) * HDc;
  const size_t base = tok * (size_t)nH * HDc + (size_t)hh * HDc;
  const float x1 = in[base + d], x2 = in[base + d + 64];
  float o1 = x1 * crow[d] - x2 * srow[d];
  float o2 = x2 * crow[d + 64] + x1 * srow[d + 64];
  if (SCALE) { o1 *= QSCALE; o2 *= QSCALE; }
  out[base + d] = (__bf16)o1;
  out[base + d + 64] = (__bf16)o2;
}

// ---------------------------------------------------------------------------
// Flash-style causal GQA attention.
// Grid (K/128, H, B); 8 waves, wave w owns q rows [qblk*128+w*16, +16).
// K (token-major) and V (pre-transposed [hd][tok]) staged via async copies.
// P bounced through per-wave LDS slab (C->A layout), wave-local s_wait_dscnt.
// ---------------------------------------------------------------------------
constexpr int LDK_S = HDc + 8;  // 136
constexpr int LDV_S = 32 + 8;   // 40
constexpr int LDP_S = 32 + 8;   // 40

__global__ __launch_bounds__(256)
void attention_kernel(const __bf16* __restrict__ qb, const __bf16* __restrict__ kb,
                      const __bf16* __restrict__ vT, __bf16* __restrict__ ob) {
  __shared__ __bf16 Ks[32 * LDK_S];
  __shared__ __bf16 Vs[HDc * LDV_S];
  __shared__ __bf16 Ps[8 * 16 * LDP_S];
  const int qblk = blockIdx.x, h = blockIdx.y, b = blockIdx.z;
  const int kvh = h >> 2;  // H/KVH = 4
  const int tid = threadIdx.x, lane = tid & 31, wid = tid >> 5;
  const int half = lane >> 4, l15 = lane & 15;
  const int q0 = qblk * 128 + wid * 16;

  v16bf qf[4];
  {
    const __bf16* qrow = qb + ((size_t)(b * Kc + q0 + l15) * Hc + h) * HDc;
#pragma unroll
    for (int kc = 0; kc < 4; ++kc) {
      const __bf16* p = qrow + kc * 32;
      qf[kc] = pack16(*(const v8bf*)(p + half * 8), *(const v8bf*)(p + 16 + half * 8));
    }
  }
  v8f O[8] = {};
  float mrow[8], lrow[8];
#pragma unroll
  for (int r = 0; r < 8; ++r) { mrow[r] = -1e30f; lrow[r] = 0.f; }

  const int jend = qblk * 128 + 128;  // uniform trip count per block
  for (int j0 = 0; j0 < jend; j0 += 32) {
    {  // stage K block (row-major [tok][hd]) : 2 async per thread
      const int rr = tid >> 3, cc = (tid & 7) * 16;
      const __bf16* src = kb + ((size_t)(b * Kc + j0 + rr) * KVHc + kvh) * HDc + cc;
      cp_async16(src, &Ks[rr * LDK_S + cc]);
      cp_async16(src + 8, &Ks[rr * LDK_S + cc + 8]);
    }
    {  // stage V block [hd][tok] straight from pre-transposed V : 2 async
      const int hd = tid >> 1, cc = (tid & 1) * 16;
      const __bf16* src = vT + ((size_t)((b * KVHc + kvh) * HDc + hd)) * Kc + j0 + cc;
      cp_async16(src, &Vs[hd * LDV_S + cc]);
      cp_async16(src + 8, &Vs[hd * LDV_S + cc + 8]);
    }
    wait_async<0>();
    __syncthreads();
    if (j0 <= q0 + 15) {  // wave-uniform: EXEC all-ones inside
      v8f s[2] = {};
#pragma unroll
      for (int kc = 0; kc < 4; ++kc)
#pragma unroll
        for (int nt = 0; nt < 2; ++nt) {
          const __bf16* p = &Ks[(nt * 16 + l15) * LDK_S + kc * 32];
          v16bf kf = pack16(*(const v8bf*)(p + half * 8), *(const v8bf*)(p + 16 + half * 8));
          s[nt] = __builtin_amdgcn_wmma_f32_16x16x32_bf16(
              false, qf[kc], false, kf, (short)0, s[nt], false, false);
        }
      // causal mask + online softmax (row = q0 + half*8 + r, col across 16 lanes)
      float mnew[8];
#pragma unroll
      for (int r = 0; r < 8; ++r) mnew[r] = mrow[r];
#pragma unroll
      for (int nt = 0; nt < 2; ++nt)
#pragma unroll
        for (int r = 0; r < 8; ++r) {
          const int row = q0 + half * 8 + r, col = j0 + nt * 16 + l15;
          float sv = (col <= row) ? s[nt][r] : -1e30f;
          s[nt][r] = sv;
          mnew[r] = fmaxf(mnew[r], sv);
        }
#pragma unroll
      for (int off = 1; off < 16; off <<= 1)
#pragma unroll
        for (int r = 0; r < 8; ++r) mnew[r] = fmaxf(mnew[r], __shfl_xor(mnew[r], off, 32));
      float scl[8], psum[8];
#pragma unroll
      for (int r = 0; r < 8; ++r) {
        scl[r] = __expf(mrow[r] - mnew[r]);
        mrow[r] = mnew[r];
        psum[r] = 0.f;
      }
#pragma unroll
      for (int nt = 0; nt < 2; ++nt)
#pragma unroll
        for (int r = 0; r < 8; ++r) {
          float sv = s[nt][r];
          float pv = (sv > -1e29f) ? __expf(sv - mnew[r]) : 0.f;
          s[nt][r] = pv;
          psum[r] += pv;
        }
#pragma unroll
      for (int off = 1; off < 16; off <<= 1)
#pragma unroll
        for (int r = 0; r < 8; ++r) psum[r] += __shfl_xor(psum[r], off, 32);
#pragma unroll
      for (int r = 0; r < 8; ++r) lrow[r] = lrow[r] * scl[r] + psum[r];
      // P (16x32) through per-wave LDS slab: C-layout -> A-fragment layout
      __bf16* myP = &Ps[wid * 16 * LDP_S];
#pragma unroll
      for (int nt = 0; nt < 2; ++nt)
#pragma unroll
        for (int r = 0; r < 8; ++r)
          myP[(half * 8 + r) * LDP_S + nt * 16 + l15] = (__bf16)s[nt][r];
#if defined(__HIP_DEVICE_COMPILE__)
      asm volatile("s_wait_dscnt 0" ::: "memory");  // wave-local LDS RAW
#endif
      v16bf pf;
      {
        const __bf16* p = &myP[l15 * LDP_S];
        pf = pack16(*(const v8bf*)(p + half * 8), *(const v8bf*)(p + 16 + half * 8));
      }
#pragma unroll
      for (int nt2 = 0; nt2 < 8; ++nt2) {
#pragma unroll
        for (int r = 0; r < 8; ++r) O[nt2][r] *= scl[r];
        const __bf16* p = &Vs[(nt2 * 16 + l15) * LDV_S];
        v16bf vf = pack16(*(const v8bf*)(p + half * 8), *(const v8bf*)(p + 16 + half * 8));
        O[nt2] = __builtin_amdgcn_wmma_f32_16x16x32_bf16(
            false, pf, false, vf, (short)0, O[nt2], false, false);
      }
    }
    __syncthreads();
  }
#pragma unroll
  for (int nt2 = 0; nt2 < 8; ++nt2)
#pragma unroll
    for (int r = 0; r < 8; ++r) {
      const int row = q0 + half * 8 + r, col = nt2 * 16 + l15;
      ob[((size_t)(b * Kc + row) * Hc + h) * HDc + col] = (__bf16)(O[nt2][r] / lrow[r]);
    }
}

// ---------------------------------------------------------------------------
// Gated scatter: out[b, idx[b,i], :] = sel + (proc - sel) * gate
// ---------------------------------------------------------------------------
__global__ __launch_bounds__(256)
void scatter_kernel(const float* __restrict__ sel, const float* __restrict__ proc,
                    const float* __restrict__ gates, const long long* __restrict__ idx,
                    float* __restrict__ out) {
  const int b = blockIdx.y, i = blockIdx.x, tid = threadIdx.x;
  const long long t = idx[b * Kc + i];
  const float g = gates[b * Kc + i];
  const size_t src = (size_t)(b * Kc + i) * Dc;
  const size_t dst = ((size_t)b * Tc + (size_t)t) * Dc;
#pragma unroll
  for (int j = 0; j < 8; ++j) {
    const int c = tid + j * 256;
    const float s = sel[src + c], p = proc[src + c];
    out[dst + c] = s + (p - s) * g;
  }
}

// ---------------------------------------------------------------------------
// Host orchestration
// ---------------------------------------------------------------------------
extern "C" void kernel_launch(void* const* d_in, const int* in_sizes, int n_in,
                              void* d_out, int out_size, void* d_ws, size_t ws_size,
                              hipStream_t stream) {
  (void)in_sizes; (void)n_in; (void)out_size; (void)ws_size;
  const float* hidden = (const float*)d_in[0];
  const long long* idx = (const long long*)d_in[1];
  const float* gates = (const float*)d_in[2];
  const float* cosT = (const float*)d_in[3];
  const float* sinT = (const float*)d_in[4];
  const float* Wq = (const float*)d_in[5];
  const float* bq = (const float*)d_in[6];
  const float* Wk = (const float*)d_in[7];
  const float* bk = (const float*)d_in[8];
  const float* Wv = (const float*)d_in[9];
  const float* bv = (const float*)d_in[10];
  const float* Wo = (const float*)d_in[11];
  const float* Wg = (const float*)d_in[12];
  const float* Wu = (const float*)d_in[13];
  const float* Wd = (const float*)d_in[14];
  const float* ln1 = (const float*)d_in[15];
  const float* ln2 = (const float*)d_in[16];

  char* ws = (char*)d_ws;
  size_t off = 0;
  auto alloc = [&](size_t bytes) -> char* {
    char* p = ws + off;
    off += (bytes + 255) & ~(size_t)255;
    return p;
  };
  float*  sel   = (float*) alloc((size_t)NTOK * Dc * 4);
  __bf16* hbf   = (__bf16*)alloc((size_t)NTOK * Dc * 2);
  __bf16* WqT   = (__bf16*)alloc((size_t)Dc * (Hc * HDc) * 2);   // [N][K]
  __bf16* WkT   = (__bf16*)alloc((size_t)Dc * (KVHc * HDc) * 2);
  __bf16* WvT   = (__bf16*)alloc((size_t)Dc * (KVHc * HDc) * 2);
  __bf16* WoT   = (__bf16*)alloc((size_t)(Hc * HDc) * Dc * 2);
  __bf16* WgT   = (__bf16*)alloc((size_t)Dc * FFc * 2);
  __bf16* WuT   = (__bf16*)alloc((size_t)Dc * FFc * 2);
  __bf16* WdT   = (__bf16*)alloc((size_t)FFc * Dc * 2);
  float*  qf32  = (float*) alloc((size_t)NTOK * Hc * HDc * 4);
  float*  kf32  = (float*) alloc((size_t)NTOK * KVHc * HDc * 4);
  __bf16* qbf   = (__bf16*)alloc((size_t)NTOK * Hc * HDc * 2);
  __bf16* kbf   = (__bf16*)alloc((size_t)NTOK * KVHc * HDc * 2);
  __bf16* vbfT  = (__bf16*)alloc((size_t)NTOK * KVHc * HDc * 2); // [b][kvh*HD][tok]
  __bf16* aobf  = (__bf16*)alloc((size_t)NTOK * Dc * 2);
  float*  xf32  = (float*) alloc((size_t)NTOK * Dc * 4);
  __bf16* h2bf  = (__bf16*)alloc((size_t)NTOK * Dc * 2);
  float*  gf32  = (float*) alloc((size_t)NTOK * FFc * 4);
  __bf16* actbf = (__bf16*)alloc((size_t)NTOK * FFc * 2);

  auto cvtT = [&](const float* src, __bf16* dst, int Kd, int N) {
    f2bT_kernel<<<dim3(N / 32, Kd / 32), 256, 0, stream>>>(src, dst, Kd, N);
  };
  cvtT(Wq, WqT, Dc, Hc * HDc);
  cvtT(Wk, WkT, Dc, KVHc * HDc);
  cvtT(Wv, WvT, Dc, KVHc * HDc);
  cvtT(Wo, WoT, Hc * HDc, Dc);
  cvtT(Wg, WgT, Dc, FFc);
  cvtT(Wu, WuT, Dc, FFc);
  cvtT(Wd, WdT, FFc, Dc);

  // gather + RMSNorm1
  rmsnorm_kernel<true><<<dim3(Kc, Bc), 256, 0, stream>>>(hidden, idx, ln1, sel, hbf);

  // QKV projections (V written transposed per (b,kvh) for attention staging)
  gemm_bf16<1><<<dim3((Hc * HDc) / GBN, NTOK / GBM), 256, 0, stream>>>(
      hbf, WqT, Hc * HDc, Dc, bq, nullptr, nullptr, qf32, nullptr);
  gemm_bf16<1><<<dim3((KVHc * HDc) / GBN, NTOK / GBM), 256, 0, stream>>>(
      hbf, WkT, KVHc * HDc, Dc, bk, nullptr, nullptr, kf32, nullptr);
  gemm_bf16<6><<<dim3((KVHc * HDc) / GBN, NTOK / GBM), 256, 0, stream>>>(
      hbf, WvT, KVHc * HDc, Dc, bv, nullptr, nullptr, nullptr, vbfT);

  // RoPE (Q scaled by 1/sqrt(HD))
  rope_kernel<true><<<dim3((unsigned)(((size_t)NTOK * Hc * 64) / 256)), 256, 0, stream>>>(
      qf32, cosT, sinT, idx, qbf, Hc);
  rope_kernel<false><<<dim3((unsigned)(((size_t)NTOK * KVHc * 64) / 256)), 256, 0, stream>>>(
      kf32, cosT, sinT, idx, kbf, KVHc);

  // causal GQA attention
  attention_kernel<<<dim3(Kc / 128, Hc, Bc), 256, 0, stream>>>(qbf, kbf, vbfT, aobf);

  // output projection + residual(selected)
  gemm_bf16<4><<<dim3(Dc / GBN, NTOK / GBM), 256, 0, stream>>>(
      aobf, WoT, Dc, Hc * HDc, nullptr, sel, nullptr, xf32, nullptr);

  // RMSNorm2
  rmsnorm_kernel<false><<<dim3(Kc, Bc), 256, 0, stream>>>(xf32, nullptr, ln2, nullptr, h2bf);

  // SwiGLU MLP
  gemm_bf16<0><<<dim3(FFc / GBN, NTOK / GBM), 256, 0, stream>>>(
      h2bf, WgT, FFc, Dc, nullptr, nullptr, nullptr, gf32, nullptr);
  gemm_bf16<5><<<dim3(FFc / GBN, NTOK / GBM), 256, 0, stream>>>(
      h2bf, WuT, FFc, Dc, nullptr, nullptr, gf32, nullptr, actbf);
  gemm_bf16<4><<<dim3(Dc / GBN, NTOK / GBM), 256, 0, stream>>>(
      actbf, WdT, Dc, FFc, nullptr, xf32, nullptr, xf32, nullptr);

  // final: copy hidden, then gated scatter of processed rows
  (void)hipMemcpyAsync(d_out, (const void*)hidden, (size_t)Bc * Tc * Dc * sizeof(float),
                       hipMemcpyDeviceToDevice, stream);
  scatter_kernel<<<dim3(Kc, Bc), 256, 0, stream>>>(sel, xf32, gates, idx, (float*)d_out);
}
